// BiSDA_3977139716506
// MI455X (gfx1250) — matrix operand order repaired and checked by reading, not compiled, not run on hardware
//
#include <hip/hip_runtime.h>

#define TT 4
#define BB 2
#define CC 128
#define SS 8192                       // 8*32*32 spatial
#define NBCS (BB * CC * SS)           // 2097152  (one time-slice)
#define NTOT ((long)TT * NBCS)        // 8388608  (full tensor)

typedef __attribute__((ext_vector_type(16))) __bf16 v16bf;
typedef __attribute__((ext_vector_type(8)))  float  v8f;

union Frag {
  v16bf v;
  uint4 q[2];
};

__device__ __forceinline__ unsigned short f2bf(float f) {
  unsigned u = __float_as_uint(f);
  unsigned r = (u + 0x7FFFu + ((u >> 16) & 1u)) >> 16;  // round-to-nearest-even
  return (unsigned short)r;
}

// window w=(dt,dh,dw) in 2x2x2, s=(slt,slh,slw) in 4x16x16 -> flat spatial
__device__ __forceinline__ int win_spatial(int w, int s) {
  int dt = (w >> 2) & 1, dh = (w >> 1) & 1, dw = w & 1;
  int slt = s >> 8, slh = (s >> 4) & 15, slw = s & 15;
  int d = dt * 4 + slt, h = dh * 16 + slh, x = dw * 16 + slw;
  return (d * 32 + h) * 32 + x;
}

// async copy of 16 bytes global -> LDS (ASYNCcnt-tracked path)
__device__ __forceinline__ void async_b128(const unsigned short* g, unsigned lds_byte_off) {
  asm volatile("global_load_async_to_lds_b128 %0, %1, off"
               :: "v"(lds_byte_off), "v"(g) : "memory");
}
__device__ __forceinline__ void wait_async0() {
  asm volatile("s_wait_asynccnt 0x0" ::: "memory");
}

// ---------------- one-time conversions -----------------------------------
__global__ void cvt_bf_kernel(const float* __restrict__ src,
                              unsigned short* __restrict__ dst, long n) {
  long i = (long)blockIdx.x * 256 + threadIdx.x;
  if (i < n) dst[i] = f2bf(src[i]);
}

// qwT[tap][o][i]: taps 0..26 conv weights tap-major (contiguous A tiles);
// tap 27 is the folded central-difference matrix -theta*kd.
__global__ void qwt_kernel(const float* __restrict__ qw,
                           unsigned short* __restrict__ qwT) {
  int i = blockIdx.x * 256 + threadIdx.x;  // 28*16384
  if (i >= 28 * 16384) return;
  int tap = i >> 14, oi = i & 16383;
  const float* p = qw + (long)oi * 27;
  float v;
  if (tap < 27) {
    v = p[tap];
  } else {
    float s = 0.f;
#pragma unroll
    for (int j = 0; j < 9; ++j) s += p[j] + p[18 + j];
    v = -0.7f * s;
  }
  qwT[i] = f2bf(v);
}

// ---------------- channel-mix GEMM (bf16 operands, f32 accum) -------------
// Y[tb][o][s] = sum_i W[o][i] * X[tb][i][s];  grid (64 s-tiles, 8 tb), 256 thr
__global__ void __launch_bounds__(256, 1)
gemm_cm_bf(const unsigned short* __restrict__ Wbf,
           const unsigned short* __restrict__ Xbf,
           float* __restrict__ Y) {
  __shared__ unsigned short lA[128 * 32];  // [m][k] bf16
  __shared__ unsigned short lB[128 * 32];  // [n][k] bf16
  const int tid = threadIdx.x;
  const int wid = tid >> 5, lane = tid & 31;
  const int s0 = blockIdx.x * 128;
  const long base = (long)blockIdx.y * CC * SS;
  const unsigned short* Xb = Xbf + base;
  float* Yb = Y + base;

  // loop-invariant staging bases (one VGPR pair + immediate offsets)
  const unsigned short* gA0 = Wbf + (tid >> 2) * 128 + (tid & 3) * 8;
  const unsigned aL0 = (unsigned)(size_t)&lA[(tid >> 2) * 32 + (tid & 3) * 8];
  const unsigned short* gB0 = Xb + (long)(tid >> 7) * SS + s0 + (tid & 127);
  unsigned short* sB = &lB[(tid & 127) * 32 + (tid >> 7)];

  v8f acc[8] = {};

  for (int kc = 0; kc < 4; ++kc) {
    // A tile 128x32: layout-preserving async DMA to LDS (2 x 16B per thread)
    async_b128(gA0 + kc * 32, aL0);
    async_b128(gA0 + kc * 32 + 64 * 128, aL0 + 64 * 32 * 2);
    if (kc < 3)  // prefetch next K-chunk of X (global_prefetch_b8)
      __builtin_prefetch(gB0 + (long)(kc + 1) * 32 * SS, 0, 1);
    // B tile 32x128 -> [n][k] transposed staging off a single base
    const unsigned short* gB = gB0 + (long)kc * 32 * SS;
#pragma unroll
    for (int r = 0; r < 16; ++r)
      sB[r * 2] = gB[(long)r * 2 * SS];
    wait_async0();
    __syncthreads();
    Frag a;  // 16x32 bf16: lanes 0-15 K{0-7,16-23}, lanes 16-31 K{8-15,24-31}
    {
      int row = wid * 16 + (lane & 15);
      int kb = (lane >= 16) ? 8 : 0;
      a.q[0] = *(const uint4*)&lA[row * 32 + kb];
      a.q[1] = *(const uint4*)&lA[row * 32 + kb + 16];
    }
    // rotating 2-deep B-fragment buffer: overlap ds_load with v_wmma
    const int kb2 = (lane >= 16) ? 8 : 0;
    const int cb = (lane & 15) * 32 + kb2;
    Frag bcur, bnxt;
    bcur.q[0] = *(const uint4*)&lB[cb];
    bcur.q[1] = *(const uint4*)&lB[cb + 16];
#pragma unroll
    for (int nt = 0; nt < 8; ++nt) {
      if (nt < 7) {
        bnxt.q[0] = *(const uint4*)&lB[cb + (nt + 1) * 16 * 32];
        bnxt.q[1] = *(const uint4*)&lB[cb + (nt + 1) * 16 * 32 + 16];
      }
      acc[nt] = __builtin_amdgcn_wmma_f32_16x16x32_bf16(false, a.v, false, bcur.v,
                                                        (short)0, acc[nt], false, false);
      bcur = bnxt;
    }
    __syncthreads();
  }
  const int m0 = wid * 16 + ((lane < 16) ? 0 : 8);
#pragma unroll
  for (int nt = 0; nt < 8; ++nt) {
    int col = s0 + nt * 16 + (lane & 15);
#pragma unroll
    for (int r = 0; r < 8; ++r)
      Yb[(long)(m0 + r) * SS + col] = acc[nt][r];
  }
}

// ---------------- implicit-GEMM 3x3x3 conv + folded central-diff (tap 27) --
__global__ void __launch_bounds__(256, 1)
conv_cdc_bf(const unsigned short* __restrict__ qwT,
            const unsigned short* __restrict__ Xbf,
            float* __restrict__ Y) {
  __shared__ unsigned short lA[128 * 32];
  __shared__ unsigned short lB[128 * 32];
  const int tid = threadIdx.x;
  const int wid = tid >> 5, lane = tid & 31;
  const int s0 = blockIdx.x * 128;          // tile: d fixed, 4 h-rows, full w
  const int d0 = s0 >> 10;
  const int h0 = (s0 & 1023) >> 5;
  const long base = (long)blockIdx.y * CC * SS;
  const unsigned short* Xb = Xbf + base;
  float* Yb = Y + base;

  const unsigned short* gA0 = qwT + (tid >> 2) * 128 + (tid & 3) * 8;
  const unsigned aL0 = (unsigned)(size_t)&lA[(tid >> 2) * 32 + (tid & 3) * 8];
  const int n = tid & 127, hl = n >> 5, wl = n & 31, kk0 = tid >> 7;
  unsigned short* sB = &lB[n * 32 + kk0];

  v8f acc[8] = {};

  for (int tap = 0; tap < 28; ++tap) {
    const bool cd = (tap == 27);            // virtual tap: A = -theta*kd, B = x
    const int td = cd ? 0 : tap / 9 - 1;
    const int th = cd ? 0 : (tap / 3) % 3 - 1;
    const int tw = cd ? 0 : tap % 3 - 1;
    // per-thread pad predicate and shifted base: loop-invariant over (r, kc)
    const int dd = d0 + td, hh = h0 + hl + th, ww = wl + tw;
    const bool valid = ((unsigned)dd < 8u) & ((unsigned)hh < 32u) & ((unsigned)ww < 32u);
    const unsigned short* gB0 = Xb + (long)kk0 * SS + (dd * 32 + hh) * 32 + ww;
    const unsigned short* gAt = gA0 + tap * 16384;
    for (int kc = 0; kc < 4; ++kc) {
      async_b128(gAt + kc * 32, aL0);
      async_b128(gAt + kc * 32 + 64 * 128, aL0 + 64 * 32 * 2);
      const unsigned short* gB = gB0 + (long)kc * 32 * SS;
#pragma unroll
      for (int r = 0; r < 16; ++r)
        sB[r * 2] = valid ? gB[(long)r * 2 * SS] : (unsigned short)0;
      wait_async0();
      __syncthreads();
      Frag a;
      {
        int row = wid * 16 + (lane & 15);
        int kb = (lane >= 16) ? 8 : 0;
        a.q[0] = *(const uint4*)&lA[row * 32 + kb];
        a.q[1] = *(const uint4*)&lA[row * 32 + kb + 16];
      }
      const int kb2 = (lane >= 16) ? 8 : 0;
      const int cb = (lane & 15) * 32 + kb2;
      Frag bcur, bnxt;
      bcur.q[0] = *(const uint4*)&lB[cb];
      bcur.q[1] = *(const uint4*)&lB[cb + 16];
#pragma unroll
      for (int nt = 0; nt < 8; ++nt) {
        if (nt < 7) {
          bnxt.q[0] = *(const uint4*)&lB[cb + (nt + 1) * 16 * 32];
          bnxt.q[1] = *(const uint4*)&lB[cb + (nt + 1) * 16 * 32 + 16];
        }
        acc[nt] = __builtin_amdgcn_wmma_f32_16x16x32_bf16(false, a.v, false, bcur.v,
                                                          (short)0, acc[nt], false, false);
        bcur = bnxt;
      }
      __syncthreads();
    }
  }
  const int m0 = wid * 16 + ((lane < 16) ? 0 : 8);
#pragma unroll
  for (int nt = 0; nt < 8; ++nt) {
    int col = s0 + nt * 16 + (lane & 15);
#pragma unroll
    for (int r = 0; r < 8; ++r)
      Yb[(long)(m0 + r) * SS + col] = acc[nt][r];
  }
}

// ---------------- per-channel batch stats (mean, biased var) ---------------
__global__ void stats_kernel(const float* __restrict__ src, float* __restrict__ stats) {
  __shared__ float r1[256], r2[256];
  const int c = blockIdx.x;
  float s = 0.f, s2 = 0.f;
  for (int i = threadIdx.x; i < TT * BB * SS; i += 256) {
    int tb = i >> 13, sp = i & (SS - 1);
    float v = src[((long)tb * CC + c) * SS + sp];
    s += v; s2 += v * v;
  }
  r1[threadIdx.x] = s; r2[threadIdx.x] = s2;
  __syncthreads();
  for (int o = 128; o > 0; o >>= 1) {
    if (threadIdx.x < o) { r1[threadIdx.x] += r1[threadIdx.x + o]; r2[threadIdx.x] += r2[threadIdx.x + o]; }
    __syncthreads();
  }
  if (threadIdx.x == 0) {
    const float inv = 1.f / (float)(TT * BB * SS);
    float m = r1[0] * inv;
    stats[c * 2] = m;
    stats[c * 2 + 1] = r2[0] * inv - m * m;
  }
}

__global__ void bn_apply(const float* __restrict__ src, float* __restrict__ dst,
                         const float* __restrict__ stats, const float* __restrict__ gamma,
                         const float* __restrict__ beta) {
  long i = (long)blockIdx.x * 256 + threadIdx.x;
  if (i >= NTOT) return;
  int c = (int)((i >> 13) & 127);
  float m = stats[c * 2], v = stats[c * 2 + 1];
  dst[i] = (src[i] - m) * rsqrtf(v + 1e-5f) * gamma[c] + beta[c];
}

// ---------------- multi-step LIF (tau=2, v_th=1, hard reset), in place -----
__global__ void lif_kernel(float* __restrict__ buf) {
  long i = (long)blockIdx.x * 256 + threadIdx.x;
  if (i >= NBCS) return;
  float vm = 0.f;
#pragma unroll
  for (int t = 0; t < 4; ++t) {
    float x = buf[(long)t * NBCS + i];
    vm += (x - vm) * 0.5f;
    float s = (vm >= 1.f) ? 1.f : 0.f;
    vm *= (1.f - s);
    buf[(long)t * NBCS + i] = s;
  }
}

// LIF writing bf16 spikes (exact: 0 / 0x3F80) for the bf16 p-GEMM input
__global__ void lif_bf_kernel(const float* __restrict__ src,
                              unsigned short* __restrict__ dst) {
  long i = (long)blockIdx.x * 256 + threadIdx.x;
  if (i >= NBCS) return;
  float vm = 0.f;
#pragma unroll
  for (int t = 0; t < 4; ++t) {
    float x = src[(long)t * NBCS + i];
    vm += (x - vm) * 0.5f;
    unsigned short s = (vm >= 1.f) ? (unsigned short)0x3F80 : (unsigned short)0;
    if (s) vm = 0.f;
    dst[(long)t * NBCS + i] = s;
  }
}

// ---------------- region means over (T, S_win) of post-BN q/k --------------
__global__ void region_kernel(const float* __restrict__ qr, const float* __restrict__ kr,
                              float* __restrict__ qreg, float* __restrict__ kreg) {
  __shared__ float r1[128], r2[128];
  const int c = blockIdx.x & 127;
  const int w = (blockIdx.x >> 7) & 7;
  const int b = blockIdx.x >> 10;
  float sq = 0.f, sk = 0.f;
  for (int i = threadIdx.x; i < 4096; i += 128) {
    int t = i >> 10, s = i & 1023;
    long off = (((long)t * BB + b) * CC + c) * SS + win_spatial(w, s);
    sq += qr[off]; sk += kr[off];
  }
  r1[threadIdx.x] = sq; r2[threadIdx.x] = sk;
  __syncthreads();
  for (int o = 64; o > 0; o >>= 1) {
    if (threadIdx.x < o) { r1[threadIdx.x] += r1[threadIdx.x + o]; r2[threadIdx.x] += r2[threadIdx.x + o]; }
    __syncthreads();
  }
  if (threadIdx.x == 0) {
    qreg[(b * 8 + w) * 128 + c] = r1[0] * (1.f / 4096.f);
    kreg[(b * 8 + w) * 128 + c] = r2[0] * (1.f / 4096.f);
  }
}

// ---------------- affinity + top-4 routing per (b, w) ----------------------
__global__ void topk_kernel(const float* __restrict__ qreg, const float* __restrict__ kreg,
                            int* __restrict__ idxb) {
  int t = threadIdx.x;
  if (t >= 16) return;
  int b = t >> 3, w = t & 7;
  float a[8];
  for (int vv = 0; vv < 8; ++vv) {
    float d = 0.f;
    for (int c = 0; c < 128; ++c)
      d += qreg[(b * 8 + w) * 128 + c] * kreg[(b * 8 + vv) * 128 + c];
    a[vv] = d * 0.25f;  // HEAD_DIM^-0.5
  }
  bool used[8] = {};
  for (int j = 0; j < 4; ++j) {
    float best = -3.4e38f; int bi = 0;
    for (int vv = 0; vv < 8; ++vv)
      if (!used[vv] && a[vv] > best) { best = a[vv]; bi = vv; }
    used[bi] = true;
    idxb[(b * 8 + w) * 4 + j] = bi;
  }
}

// ---------------- routed spiking attention; block = (b, w, s-chunk) --------
__global__ void attn_kernel(const float* __restrict__ q, const float* __restrict__ k,
                            const float* __restrict__ v, const int* __restrict__ idxb,
                            float* __restrict__ out) {
  const int sc = blockIdx.x & 3;
  const int w = (blockIdx.x >> 2) & 7;
  const int b = blockIdx.x >> 5;
  const int s = sc * 256 + threadIdx.x;
  const int spT = win_spatial(w, s);
  int spj[4];
#pragma unroll
  for (int j = 0; j < 4; ++j) spj[j] = win_spatial(idxb[(b * 8 + w) * 4 + j], s);

  float vst[8];
#pragma unroll
  for (int h = 0; h < 8; ++h) vst[h] = 0.f;

  for (int t = 0; t < 4; ++t) {
    const long baseT = ((long)t * BB + b) * CC;
    float spike[8];
    for (int h = 0; h < 8; ++h) {
      float acc = 0.f;
#pragma unroll
      for (int d = 0; d < 16; ++d) {
        long ro = (baseT + h * 16 + d) * SS;
        float qv = q[ro + spT];
        float ka = 0.25f * (k[ro + spj[0]] + k[ro + spj[1]] + k[ro + spj[2]] + k[ro + spj[3]]);
        acc += qv * ka;
      }
      vst[h] += (acc - vst[h]) * 0.5f;   // LIF on attn
      float sp = (vst[h] >= 1.f) ? 1.f : 0.f;
      vst[h] *= (1.f - sp);
      spike[h] = sp;
    }
    for (int h = 0; h < 8; ++h) {
#pragma unroll
      for (int d = 0; d < 16; ++d) {
        long ro = (baseT + h * 16 + d) * SS;
        float va = 0.25f * (v[ro + spj[0]] + v[ro + spj[1]] + v[ro + spj[2]] + v[ro + spj[3]]);
        out[ro + spT] = spike[h] * va;
      }
    }
  }
}

extern "C" void kernel_launch(void* const* d_in, const int* in_sizes, int n_in,
                              void* d_out, int out_size, void* d_ws, size_t ws_size,
                              hipStream_t stream) {
  (void)in_sizes; (void)n_in; (void)out_size; (void)ws_size;
  const float* x  = (const float*)d_in[0];
  const float* qw = (const float*)d_in[1];
  const float* qg = (const float*)d_in[2];
  const float* qb = (const float*)d_in[3];
  const float* kw = (const float*)d_in[4];
  const float* kg = (const float*)d_in[5];
  const float* kb = (const float*)d_in[6];
  const float* vg = (const float*)d_in[7];
  const float* vb = (const float*)d_in[8];
  const float* pw = (const float*)d_in[9];
  const float* pg = (const float*)d_in[10];
  const float* pb = (const float*)d_in[11];
  float* out = (float*)d_out;

  float* ws = (float*)d_ws;
  const long N = NTOT;
  float* bufQ = ws;               // q_pre -> q_real -> q spikes -> reused for p_pre
  float* bufK = ws + N;
  float* bufV = ws + 2 * N;
  float* bufO = ws + 3 * N;       // attn out (f32)
  unsigned short* xbf = (unsigned short*)(ws + 4 * N);          // bf16(x)
  unsigned short* obf = (unsigned short*)(ws + 4 * N + N / 2);  // bf16 out-spikes
  unsigned short* qwT = (unsigned short*)(ws + 5 * N);          // 28*16384
  unsigned short* wK  = qwT + 28 * 16384;                       // 16384
  unsigned short* wP  = wK + 16384;                             // 16384
  float* stQ  = (float*)(wP + 16384);
  float* stK  = stQ + 256;
  float* stV  = stK + 256;
  float* stP  = stV + 256;
  float* qreg = stP + 256;        // [2,8,128]
  float* kreg = qreg + 2048;
  int*   idxb = (int*)(kreg + 2048);

  dim3 gGemm(64, 8);

  // one-time bf16 materialization (halves all GEMM tile-staging traffic)
  cvt_bf_kernel<<<32768, 256, 0, stream>>>(x, xbf, NTOT);
  cvt_bf_kernel<<<64, 256, 0, stream>>>(kw, wK, 16384);
  cvt_bf_kernel<<<64, 256, 0, stream>>>(pw, wP, 16384);
  qwt_kernel<<<1792, 256, 0, stream>>>(qw, qwT);

  gemm_cm_bf<<<gGemm, 256, 0, stream>>>(wK, xbf, bufK);        // k_pre (WMMA)
  conv_cdc_bf<<<gGemm, 256, 0, stream>>>(qwT, xbf, bufQ);      // q_pre (WMMA)
  stats_kernel<<<128, 256, 0, stream>>>(bufQ, stQ);
  stats_kernel<<<128, 256, 0, stream>>>(bufK, stK);
  stats_kernel<<<128, 256, 0, stream>>>(x, stV);
  bn_apply<<<32768, 256, 0, stream>>>(bufQ, bufQ, stQ, qg, qb);
  bn_apply<<<32768, 256, 0, stream>>>(bufK, bufK, stK, kg, kb);
  bn_apply<<<32768, 256, 0, stream>>>(x, bufV, stV, vg, vb);
  region_kernel<<<2048, 128, 0, stream>>>(bufQ, bufK, qreg, kreg);
  topk_kernel<<<1, 32, 0, stream>>>(qreg, kreg, idxb);
  lif_kernel<<<8192, 256, 0, stream>>>(bufQ);
  lif_kernel<<<8192, 256, 0, stream>>>(bufK);
  lif_kernel<<<8192, 256, 0, stream>>>(bufV);
  attn_kernel<<<64, 256, 0, stream>>>(bufQ, bufK, bufV, idxb, bufO);
  lif_bf_kernel<<<8192, 256, 0, stream>>>(bufO, obf);          // spikes -> bf16
  gemm_cm_bf<<<gGemm, 256, 0, stream>>>(wP, obf, bufQ);        // p_pre (WMMA)
  stats_kernel<<<128, 256, 0, stream>>>(bufQ, stP);
  bn_apply<<<32768, 256, 0, stream>>>(bufQ, out, stP, pg, pb);
}
